// MultiHeadAttention_82171314307203
// MI455X (gfx1250) — matrix operand order repaired and checked by reading, not compiled, not run on hardware
//
#include <hip/hip_runtime.h>

#define NB 2
#define NS 2048
#define ND 1024
#define NH 16
#define NDK 64
#define KCH 128   // key-chunk per attention pass
#define NW (NS / 32)  // mask words per row

typedef __attribute__((ext_vector_type(16))) _Float16 v16h;
typedef __attribute__((ext_vector_type(8)))  _Float16 v8h;
typedef __attribute__((ext_vector_type(8)))  float    v8f;

// ---------------- WMMA fragment helpers (CDNA5 wave32 layouts) ----------------

// A 16x32 f16 fragment from row-major [16 x ld] tile at base.
// lanes 0-15: M=lane, K={0..7,16..23}; lanes 16-31: M=lane-16, K={8..15,24..31}
static __device__ __forceinline__ v16h load_a16(const _Float16* __restrict__ base, int ld) {
  const int lane = threadIdx.x & 31;
  const _Float16* p = base + (lane & 15) * ld + ((lane >> 4) << 3);
  v8h lo = *(const v8h*)(p);
  v8h hi = *(const v8h*)(p + 16);
  v16h a;
#pragma unroll
  for (int i = 0; i < 8; ++i) { a[i] = lo[i]; a[i + 8] = hi[i]; }
  return a;
}

// B 32x16 f16 fragment where element (k,n) lives at base[n*ld + k]  (NT operand:
// row-major weights [N x K]).  lane = N column; lanes 0-15: K={0..7,16..23}, lanes 16-31: K={8..15,24..31}.
static __device__ __forceinline__ v16h load_bt16(const _Float16* __restrict__ base, int ld) {
  const int lane = threadIdx.x & 31;
  const _Float16* p = base + (lane & 15) * ld + ((lane >> 4) << 3);
  v8h lo = *(const v8h*)(p);
  v8h hi = *(const v8h*)(p + 16);
  v16h b;
#pragma unroll
  for (int i = 0; i < 8; ++i) { b[i] = lo[i]; b[i + 8] = hi[i]; }
  return b;
}

static __device__ __forceinline__ v8f wmma16(v16h a, v16h b, v8f c) {
  return __builtin_amdgcn_wmma_f32_16x16x32_f16(false, a, false, b, (short)0, c, false, false);
}

// ---------------- small utility kernels ----------------

__global__ void mha_cvt_f32_f16(const float* __restrict__ s, _Float16* __restrict__ d, int n) {
  int i = blockIdx.x * blockDim.x + threadIdx.x;
  if (i < n) d[i] = (_Float16)s[i];
}

__global__ void mha_zero_f32(float* __restrict__ p, int n) {
  int i = blockIdx.x * blockDim.x + threadIdx.x;
  if (i < n) p[i] = 0.0f;
}

// pack mask [B,S,S] int32 -> [B,S,S/32] bitmask (bit=1 -> attend)
__global__ void mha_pack_mask(const int* __restrict__ mask, unsigned* __restrict__ bits, int nwords) {
  int w = blockIdx.x * blockDim.x + threadIdx.x;
  if (w >= nwords) return;
  const int* p = mask + (size_t)w * 32;
  unsigned v = 0;
#pragma unroll
  for (int i = 0; i < 32; ++i) v |= (p[i] != 0) ? (1u << i) : 0u;
  bits[w] = v;
}

// ---------------- projection GEMM:  C[M=4096,N=1024] = A[M,K]*W[N,K]^T + bias ----------------
// Each wave: 32x64 tile (2 A-frags x 4 B-frags -> 8 WMMAs per k-step), next-k
// fragments double-buffered so loads overlap WMMA.
// mode 0: f16 out, [B,H,S,DK] (Q/K heads); mode 1: f16 out, [B,H,DK,S] (V^T);
// mode 2: f32 out, row-major [M,N] (final output)
__launch_bounds__(256)
__global__ void mha_proj_gemm(const _Float16* __restrict__ A, const _Float16* __restrict__ W,
                              const float* __restrict__ bias, _Float16* __restrict__ out16,
                              float* __restrict__ out32, int mode) {
  const int wid  = (blockIdx.x * blockDim.x + threadIdx.x) >> 5;  // 0..2047
  const int lane = threadIdx.x & 31;
  const int tm = (wid >> 4) << 5;   // 32-row tile base
  const int tn = (wid & 15) << 6;   // 64-col strip base

  const _Float16* A0 = A + (size_t)tm * ND;
  const _Float16* A1 = A + (size_t)(tm + 16) * ND;
  const _Float16* W0 = W + (size_t)tn * ND;

  v8f acc[2][4] = {};
  v16h a0 = load_a16(A0, ND);
  v16h a1 = load_a16(A1, ND);
  v16h b0 = load_bt16(W0, ND);
  v16h b1 = load_bt16(W0 + (size_t)16 * ND, ND);
  v16h b2 = load_bt16(W0 + (size_t)32 * ND, ND);
  v16h b3 = load_bt16(W0 + (size_t)48 * ND, ND);

  for (int k0 = 0; k0 < ND; k0 += 32) {
    const int kn = (k0 + 32 < ND) ? (k0 + 32) : 0;   // harmless wrap on last iter
    v16h na0 = load_a16(A0 + kn, ND);
    v16h na1 = load_a16(A1 + kn, ND);
    v16h nb0 = load_bt16(W0 + kn, ND);
    v16h nb1 = load_bt16(W0 + (size_t)16 * ND + kn, ND);
    v16h nb2 = load_bt16(W0 + (size_t)32 * ND + kn, ND);
    v16h nb3 = load_bt16(W0 + (size_t)48 * ND + kn, ND);

    acc[0][0] = wmma16(a0, b0, acc[0][0]);
    acc[1][0] = wmma16(a1, b0, acc[1][0]);
    acc[0][1] = wmma16(a0, b1, acc[0][1]);
    acc[1][1] = wmma16(a1, b1, acc[1][1]);
    acc[0][2] = wmma16(a0, b2, acc[0][2]);
    acc[1][2] = wmma16(a1, b2, acc[1][2]);
    acc[0][3] = wmma16(a0, b3, acc[0][3]);
    acc[1][3] = wmma16(a1, b3, acc[1][3]);

    a0 = na0; a1 = na1; b0 = nb0; b1 = nb1; b2 = nb2; b3 = nb3;
  }

  const int col = lane & 15;
  const int ro  = (lane >> 4) << 3;
#pragma unroll
  for (int i = 0; i < 2; ++i) {
#pragma unroll
    for (int j = 0; j < 4; ++j) {
      const int n = tn + j * 16 + col;
      const float bv = bias[n];
#pragma unroll
      for (int r = 0; r < 8; ++r) {
        const int m = tm + i * 16 + ro + r;
        const float v = acc[i][j][r] + bv;
        if (mode == 2) {
          out32[(size_t)m * ND + n] = v;
        } else {
          const int b  = m >> 11;          // S = 2048
          const int q  = m & (NS - 1);
          const int h  = n >> 6;           // DK = 64
          const int dk = n & (NDK - 1);
          if (mode == 0)
            out16[(((size_t)(b * NH + h)) * NS + q) * NDK + dk] = (_Float16)v;
          else
            out16[(((size_t)(b * NH + h)) * NDK + dk) * NS + q] = (_Float16)v;
        }
      }
    }
  }
}

// ---------------- attention: one block = (b,h, 16-query tile) ----------------
__launch_bounds__(256)
__global__ void mha_attention(const _Float16* __restrict__ Qh, const _Float16* __restrict__ Kh,
                              const _Float16* __restrict__ Vt, const unsigned* __restrict__ mbits,
                              _Float16* __restrict__ ctx, float* __restrict__ avg) {
  const int qt = blockIdx.x & (NS / 16 - 1);   // 128 query tiles
  const int bh = blockIdx.x >> 7;              // 0..31
  const int h  = bh & (NH - 1);
  const int b  = bh >> 4;
  const int q0 = qt << 4;

  const _Float16* Qb   = Qh + ((size_t)bh * NS + q0) * NDK;  // [16][64]
  const _Float16* Kb   = Kh + (size_t)bh * NS * NDK;         // [2048][64]
  const _Float16* Vb   = Vt + (size_t)bh * NDK * NS;         // [64][2048]
  float*          avgb = avg + ((size_t)b * NS + q0) * NS;

  __shared__ float    sc[16][KCH];
  __shared__ _Float16 pp[16][KCH];
  __shared__ float    red[16][16];
  __shared__ float    rowmax[16], rowsum[16], cmx[16];

  const int tid = threadIdx.x;
  const int wid = tid >> 5, lane = tid & 31;
  const int rr = tid >> 4, rs = tid & 15;      // 16 threads per score row

  // packed-mask word base for this thread's row; word covers 32 keys
  const unsigned* mrow = mbits + ((size_t)b * NS + q0 + rr) * NW;
  const int bitbase = (rs & 3) * 8;            // bit offset of this thread's 8 keys

  // per-block-constant Q fragments (K=0..31, 32..63)
  const v16h qa0 = load_a16(Qb, NDK);
  const v16h qa1 = load_a16(Qb + 32, NDK);

  if (tid < 16) { rowmax[tid] = -3.0e38f; rowsum[tid] = 0.0f; }
  __syncthreads();

  const float scale = 0.125f;  // 1/sqrt(64)

  // ---- pass 1: row max / row sum (online over key chunks) ----
  for (int kc = 0; kc < NS; kc += KCH) {
    const int cb = kc + wid * 16;   // this wave's 16 key columns
    v16h kb0 = load_bt16(Kb + (size_t)cb * NDK, NDK);
    v16h kb1 = load_bt16(Kb + (size_t)cb * NDK + 32, NDK);
    v8f s = {};
    s = wmma16(qa0, kb0, s);
    s = wmma16(qa1, kb1, s);
    {
      const int col = lane & 15, ro = (lane >> 4) << 3;
#pragma unroll
      for (int r = 0; r < 8; ++r) sc[ro + r][wid * 16 + col] = s[r] * scale;
    }
    __syncthreads();

    const unsigned mw = mrow[(kc >> 5) + (rs >> 2)];
    float mx = -3.0e38f;
#pragma unroll
    for (int c = 0; c < KCH / 16; ++c) {
      const float v = ((mw >> (bitbase + c)) & 1u) ? sc[rr][rs * (KCH / 16) + c] : -3.0e38f;
      mx = fmaxf(mx, v);
    }
    red[rr][rs] = mx;
    __syncthreads();
    if (rs == 0) {
      float m = red[rr][0];
#pragma unroll
      for (int i = 1; i < 16; ++i) m = fmaxf(m, red[rr][i]);
      cmx[rr] = m;
    }
    __syncthreads();
    const float cm = cmx[rr];
    float sm = 0.0f;
#pragma unroll
    for (int c = 0; c < KCH / 16; ++c) {
      if ((mw >> (bitbase + c)) & 1u) sm += __expf(sc[rr][rs * (KCH / 16) + c] - cm);
    }
    red[rr][rs] = sm;
    __syncthreads();
    if (rs == 0) {
      float t = 0.0f;
#pragma unroll
      for (int i = 0; i < 16; ++i) t += red[rr][i];
      const float mo = rowmax[rr];
      const float mn = fmaxf(mo, cm);
      rowsum[rr] = rowsum[rr] * __expf(mo - mn) + t * __expf(cm - mn);
      rowmax[rr] = mn;
    }
    __syncthreads();
  }

  // ---- pass 2: P = softmax, avg accumulation, context = P @ V ----
  v8f cacc = {};   // waves 0..3 each own a 16-wide dk tile of the 16x64 context
  for (int kc = 0; kc < NS; kc += KCH) {
    const int cb = kc + wid * 16;
    v16h kb0 = load_bt16(Kb + (size_t)cb * NDK, NDK);
    v16h kb1 = load_bt16(Kb + (size_t)cb * NDK + 32, NDK);
    v8f s = {};
    s = wmma16(qa0, kb0, s);
    s = wmma16(qa1, kb1, s);
    {
      const int col = lane & 15, ro = (lane >> 4) << 3;
#pragma unroll
      for (int r = 0; r < 8; ++r) sc[ro + r][wid * 16 + col] = s[r] * scale;
    }
    __syncthreads();

    const unsigned mw = mrow[(kc >> 5) + (rs >> 2)];
    const float mx  = rowmax[rr];
    const float inv = 1.0f / rowsum[rr];
#pragma unroll
    for (int c = 0; c < KCH / 16; ++c) {
      const int cc = rs * (KCH / 16) + c;
      const float p = ((mw >> (bitbase + c)) & 1u) ? __expf(sc[rr][cc] - mx) * inv : 0.0f;
      pp[rr][cc] = (_Float16)p;
      unsafeAtomicAdd(&avgb[(size_t)rr * NS + kc + cc], p * (1.0f / NH));
    }
    __syncthreads();

    if (wid < 4) {   // wave-uniform: EXEC all ones inside
#pragma unroll
      for (int kk = 0; kk < KCH; kk += 32) {
        v16h pa = load_a16(&pp[0][0] + kk, KCH);
        v16h vb = load_bt16(Vb + (size_t)(wid * 16) * NS + kc + kk, NS);
        cacc = wmma16(pa, vb, cacc);
      }
    }
    __syncthreads();
  }

  if (wid < 4) {
    const int col = lane & 15, ro = (lane >> 4) << 3;
    const int dk0 = wid * 16;
#pragma unroll
    for (int r = 0; r < 8; ++r) {
      const int q = q0 + ro + r;
      ctx[((size_t)b * NS + q) * ND + h * NDK + dk0 + col] = (_Float16)cacc[r];
    }
  }
}

// ---------------- host launcher ----------------
extern "C" void kernel_launch(void* const* d_in, const int* in_sizes, int n_in,
                              void* d_out, int out_size, void* d_ws, size_t ws_size,
                              hipStream_t stream) {
  const float* q    = (const float*)d_in[0];
  const float* k    = (const float*)d_in[1];
  const float* v    = (const float*)d_in[2];
  const int*   mask = (const int*)d_in[3];
  const float* Wq_w = (const float*)d_in[4];
  const float* Wq_b = (const float*)d_in[5];
  const float* Wk_w = (const float*)d_in[6];
  const float* Wk_b = (const float*)d_in[7];
  const float* Wv_w = (const float*)d_in[8];
  const float* Wv_b = (const float*)d_in[9];
  const float* Wo_w = (const float*)d_in[10];
  const float* Wo_b = (const float*)d_in[11];

  float* out = (float*)d_out;                       // [B,S,D]
  float* avg = out + (size_t)NB * NS * ND;          // [B,S,S]

  const size_t SZ_BSD = (size_t)NB * NS * ND;       // 4194304
  const size_t SZ_DD  = (size_t)ND * ND;            // 1048576

  _Float16* q16   = (_Float16*)d_ws;
  _Float16* k16   = q16 + SZ_BSD;
  _Float16* vv16  = k16 + SZ_BSD;
  _Float16* wq16  = vv16 + SZ_BSD;
  _Float16* wk16  = wq16 + SZ_DD;
  _Float16* wv16  = wk16 + SZ_DD;
  _Float16* wo16  = wv16 + SZ_DD;
  _Float16* Qh    = wo16 + SZ_DD;
  _Float16* Kh    = Qh + SZ_BSD;
  _Float16* Vt    = Kh + SZ_BSD;
  _Float16* ctx   = Vt + SZ_BSD;
  unsigned* mbits = (unsigned*)(ctx + SZ_BSD);      // [B,S,S/32]

  auto cvt = [&](const float* s, _Float16* d, size_t n) {
    mha_cvt_f32_f16<<<(int)((n + 255) / 256), 256, 0, stream>>>(s, d, (int)n);
  };
  cvt(q, q16, SZ_BSD);
  cvt(k, k16, SZ_BSD);
  cvt(v, vv16, SZ_BSD);
  cvt(Wq_w, wq16, SZ_DD);
  cvt(Wk_w, wk16, SZ_DD);
  cvt(Wv_w, wv16, SZ_DD);
  cvt(Wo_w, wo16, SZ_DD);

  const int nwords = NB * NS * NW;                  // 262144
  mha_pack_mask<<<(nwords + 255) / 256, 256, 0, stream>>>(mask, mbits, nwords);

  const size_t navg = (size_t)NB * NS * NS;
  mha_zero_f32<<<(int)((navg + 255) / 256), 256, 0, stream>>>(avg, (int)navg);

  // 2048 wave-tiles (128 32-row tiles x 16 strips) / 8 waves per block = 256 blocks
  mha_proj_gemm<<<256, 256, 0, stream>>>(q16,  wq16, Wq_b, Qh, nullptr, 0);
  mha_proj_gemm<<<256, 256, 0, stream>>>(k16,  wk16, Wk_b, Kh, nullptr, 0);
  mha_proj_gemm<<<256, 256, 0, stream>>>(vv16, wv16, Wv_b, Vt, nullptr, 1);

  mha_attention<<<NB * NH * (NS / 16), 256, 0, stream>>>(Qh, Kh, Vt, mbits, ctx, avg);

  mha_proj_gemm<<<256, 256, 0, stream>>>(ctx, wo16, Wo_b, nullptr, out, 2);
}
